// TIMTremorSNN_68186900791560
// MI455X (gfx1250) — matrix operand order repaired and checked by reading, not compiled, not run on hardware
//
#include <hip/hip_runtime.h>

typedef __attribute__((ext_vector_type(16))) _Float16 v16h;
typedef __attribute__((ext_vector_type(8)))  float    v8f;

#define BETA 0.95f
#define THR  1.0f

#define BATCH   8192
#define TILE_M  16
#define WAVES   4          // waves (16-batch tiles) per block
#define ROWH    136        // halves per stage row (padded: avoids LDS bank conflicts)
#define NTILES  29         // total 32x16 f16 weight fragments
#define TILEH   512        // halves per weight tile in LDS (32 lanes * 16 halves)

// ---------------- WMMA helpers ----------------

__device__ __forceinline__ v8f wmma_f16(v16h a, v16h b, v8f c) {
  // D = A(16x32 f16) x B(32x16 f16) + C(16x16 f32)
  return __builtin_amdgcn_wmma_f32_16x16x32_f16(false, a, false, b, (short)0, c,
                                                false, false);
}

__device__ __forceinline__ v8f bias_frag(float b) {
  v8f c;
#pragma unroll
  for (int i = 0; i < 8; ++i) c[i] = b;
  return c;
}

// LIF: reset from OLD membrane (stop-gradient detach irrelevant for fwd),
// mem = beta*mem + cur - reset*THR ; spk = (mem - THR) > 0
__device__ __forceinline__ v8f lif_frag(v8f& mem, v8f acc) {
  v8f spk;
#pragma unroll
  for (int i = 0; i < 8; ++i) {
    float reset = (mem[i] > THR) ? THR : 0.0f;
    float m = BETA * mem[i] + acc[i] - reset;
    mem[i] = m;
    spk[i] = (m > THR) ? 1.0f : 0.0f;
  }
  return spk;
}

// Load a B fragment: pre-swizzled, 32 contiguous bytes per lane -> 2x ds_load_b128
__device__ __forceinline__ v16h ldB(const _Float16* wlds, int tile, int lane) {
  return *(const v16h*)&wlds[tile * TILEH + lane * 16];
}

// Load an A fragment (16-bit 16x32 layout) from the stage buffer.
// element(m,k): m = lane%16 ; k = (v<4 ? 2v : 2v+8) + 8*(lane>=16) + h
__device__ __forceinline__ v16h ldA(const _Float16* srow, int m, int hi, int koff) {
  union { v16h v; unsigned u[8]; } r;
#pragma unroll
  for (int vi = 0; vi < 8; ++vi) {
    int kb = (vi < 4 ? 2 * vi : 2 * vi + 8) + hi * 8 + koff;  // even -> 4B aligned
    r.u[vi] = *(const unsigned*)(srow + m * ROWH + kb);
  }
  return r.v;
}

// Store a spike D fragment (C/D layout: vgpr i -> row m=i+8*hi, col n=lane%16)
__device__ __forceinline__ void stS(_Float16* srow, int n, int hi, int koff, v8f spk) {
#pragma unroll
  for (int i = 0; i < 8; ++i)
    srow[(i + 8 * hi) * ROWH + koff + n] = (_Float16)spk[i];
}

// Layer-1 fused input feature k of [left(6) | right(6) | ctx(3) | 0...]
__device__ __forceinline__ float in_feat(int k, const float* l6, const float* r6,
                                         const float* c3) {
  if (k < 6)  return l6[k];
  if (k < 12) return r6[k - 6];
  if (k < 15) return c3[k - 12];
  return 0.0f;
}

// Weight element for fragment tile `tile`, local (k in 0..31, n in 0..15).
// Tiles: 0..4  block-diag layer1 [k:0..14 -> l|r|c, n:0..79 -> l1|r1|c1]
//        5..6  Wl2  7..8 Wr2
//        9..20 Wf  (t = 9 + nc*3 + kc, nc 0..3, kc 0..2, fan_in 80)
//        21..26 Wf2 (t = 21 + nc*2 + kc)
//        27..28 Wo  (kc 0..1, fan_out 4, fan_in 48)
__device__ __forceinline__ float wval(int tile, int k, int nl,
    const float* Wl1, const float* Wl2, const float* Wr1, const float* Wr2,
    const float* Wc, const float* Wf, const float* Wf2, const float* Wo) {
  if (tile < 5) {
    int ng = tile * 16 + nl;
    if (ng < 32)      return (k < 6)             ? Wl1[ng * 6 + k]              : 0.0f;
    else if (ng < 64) return (k >= 6 && k < 12)  ? Wr1[(ng - 32) * 6 + k - 6]   : 0.0f;
    else              return (k >= 12 && k < 15) ? Wc[(ng - 64) * 3 + k - 12]   : 0.0f;
  }
  if (tile < 7)  return Wl2[((tile - 5) * 16 + nl) * 32 + k];
  if (tile < 9)  return Wr2[((tile - 7) * 16 + nl) * 32 + k];
  if (tile < 21) { int tt = tile - 9;  int nc = tt / 3, kc = tt % 3;
                   int kg = kc * 32 + k;
                   return (kg < 80) ? Wf[(nc * 16 + nl) * 80 + kg] : 0.0f; }
  if (tile < 27) { int tt = tile - 21; int nc = tt / 2, kc = tt % 2;
                   return Wf2[(nc * 16 + nl) * 64 + kc * 32 + k]; }
  { int kg = (tile - 27) * 32 + k;
    return (nl < 4 && kg < 48) ? Wo[nl * 48 + kg] : 0.0f; }
}

__global__ __launch_bounds__(WAVES * 32)
void snn_lif_wmma_kernel(
    const float* __restrict__ L,  const float* __restrict__ R,
    const float* __restrict__ Cx,
    const float* __restrict__ Wl1, const float* __restrict__ bl1,
    const float* __restrict__ Wl2, const float* __restrict__ bl2,
    const float* __restrict__ Wr1, const float* __restrict__ br1,
    const float* __restrict__ Wr2, const float* __restrict__ br2,
    const float* __restrict__ Wc,  const float* __restrict__ bc,
    const float* __restrict__ Wf,  const float* __restrict__ bf,
    const float* __restrict__ Wf2, const float* __restrict__ bf2,
    const float* __restrict__ Wo,  const float* __restrict__ bo,
    float* __restrict__ out, int T)
{
  __shared__ __align__(32) _Float16 wlds[NTILES * TILEH];        // ~29.7 KB
  __shared__ __align__(32) _Float16 stage[WAVES * 16 * ROWH];    // ~17.4 KB

  const int tid  = threadIdx.x;
  const int lane = tid & 31;
  const int wv   = tid >> 5;
  const int n    = lane & 15;          // col (C/D) / row (A) / col (B)
  const int hi   = (lane >> 4) & 1;    // lane-half selector
  const int m0   = (blockIdx.x * WAVES + wv) * TILE_M;

  // ---- preload weights into LDS as per-lane B fragments (f16) ----
  for (int idx = tid; idx < NTILES * 32; idx += blockDim.x) {
    int tile = idx >> 5;
    int ln   = idx & 31;
    int nl   = ln & 15;
    int kh   = (ln >> 4) * 8;          // B layout: lanes 16-31 hold K=8..15,24..31
    _Float16* dst = &wlds[tile * TILEH + ln * 16];
    for (int j = 0; j < 16; ++j)
      dst[j] = (_Float16)wval(tile, kh + (j < 8 ? j : j + 8), nl, Wl1, Wl2, Wr1, Wr2, Wc, Wf, Wf2, Wo);
  }
  // zero the stage so padded K-slots are exact 0 (never NaN bits into WMMA)
  for (int idx = tid; idx < WAVES * 16 * ROWH; idx += blockDim.x)
    stage[idx] = (_Float16)0.0f;
  __syncthreads();

  _Float16* srow = &stage[wv * 16 * ROWH];

  // per-lane bias values (C/D col n = lane%16)
  const float b_l1a = bl1[n],     b_l1b = bl1[16 + n];
  const float b_r1a = br1[n],     b_r1b = br1[16 + n];
  const float b_c1  = bc[n];
  const float b_l2a = bl2[n],     b_l2b = bl2[16 + n];
  const float b_r2a = br2[n],     b_r2b = br2[16 + n];
  const float b_fv[4] = { bf[n], bf[16 + n], bf[32 + n], bf[48 + n] };
  const float b_gv[3] = { bf2[n], bf2[16 + n], bf2[32 + n] };
  const float b_o = (n < 4) ? bo[n] : 0.0f;

  // membrane state: 17 C/D fragments resident in VGPRs
  v8f m_l1a = {}, m_l1b = {}, m_r1a = {}, m_r1b = {}, m_c1 = {};
  v8f m_l2a = {}, m_l2b = {}, m_r2a = {}, m_r2b = {};
  v8f m_f[4] = {{}, {}, {}, {}};
  v8f m_g[3] = {{}, {}, {}};
  v8f m_o = {};

  for (int t = 0; t < T; ++t) {
    const size_t row = (size_t)t * BATCH + (size_t)(m0 + n);
    const float* l6 = L  + row * 6;
    const float* r6 = R  + row * 6;
    const float* c3 = Cx + row * 3;

    // ---- phase 1: inputs -> fused layer-1 A fragment (15 real features) ----
    union { v16h v; _Float16 e[16]; } a1;
#pragma unroll
    for (int vi = 0; vi < 8; ++vi) {
      int kb = (vi < 4 ? 2 * vi : 2 * vi + 8) + hi * 8;
      a1.e[2 * vi]     = (_Float16)in_feat(kb,     l6, r6, c3);
      a1.e[2 * vi + 1] = (_Float16)in_feat(kb + 1, l6, r6, c3);
    }
    v8f acc;
    acc = wmma_f16(a1.v, ldB(wlds, 0, lane), bias_frag(b_l1a)); v8f s_l1a = lif_frag(m_l1a, acc);
    acc = wmma_f16(a1.v, ldB(wlds, 1, lane), bias_frag(b_l1b)); v8f s_l1b = lif_frag(m_l1b, acc);
    acc = wmma_f16(a1.v, ldB(wlds, 2, lane), bias_frag(b_r1a)); v8f s_r1a = lif_frag(m_r1a, acc);
    acc = wmma_f16(a1.v, ldB(wlds, 3, lane), bias_frag(b_r1b)); v8f s_r1b = lif_frag(m_r1b, acc);
    acc = wmma_f16(a1.v, ldB(wlds, 4, lane), bias_frag(b_c1));  v8f s_c1  = lif_frag(m_c1,  acc);
    stS(srow, n, hi, 0,  s_l1a); stS(srow, n, hi, 16, s_l1b);
    stS(srow, n, hi, 32, s_r1a); stS(srow, n, hi, 48, s_r1b);
    stS(srow, n, hi, 64, s_c1);      // s_c parked at k=64..79 until fused layer
    __syncthreads();

    // ---- phase 2: layer 2 (left/right 32->32) ----
    v16h a2l = ldA(srow, n, hi, 0);
    v16h a2r = ldA(srow, n, hi, 32);
    acc = wmma_f16(a2l, ldB(wlds, 5, lane), bias_frag(b_l2a)); v8f s_l2a = lif_frag(m_l2a, acc);
    acc = wmma_f16(a2l, ldB(wlds, 6, lane), bias_frag(b_l2b)); v8f s_l2b = lif_frag(m_l2b, acc);
    acc = wmma_f16(a2r, ldB(wlds, 7, lane), bias_frag(b_r2a)); v8f s_r2a = lif_frag(m_r2a, acc);
    acc = wmma_f16(a2r, ldB(wlds, 8, lane), bias_frag(b_r2b)); v8f s_r2b = lif_frag(m_r2b, acc);
    __syncthreads();
    stS(srow, n, hi, 0,  s_l2a); stS(srow, n, hi, 16, s_l2b);
    stS(srow, n, hi, 32, s_r2a); stS(srow, n, hi, 48, s_r2b);
    __syncthreads();

    // ---- phase 3: fused 80 -> 64 (K chunks 0..2; k>=80 weights are zero) ----
    v16h af0 = ldA(srow, n, hi, 0);
    v16h af1 = ldA(srow, n, hi, 32);
    v16h af2 = ldA(srow, n, hi, 64);
    v8f sf[4];
#pragma unroll
    for (int nc = 0; nc < 4; ++nc) {
      acc = bias_frag(b_fv[nc]);
      acc = wmma_f16(af0, ldB(wlds, 9 + nc * 3 + 0, lane), acc);
      acc = wmma_f16(af1, ldB(wlds, 9 + nc * 3 + 1, lane), acc);
      acc = wmma_f16(af2, ldB(wlds, 9 + nc * 3 + 2, lane), acc);
      sf[nc] = lif_frag(m_f[nc], acc);
    }
    __syncthreads();
    stS(srow, n, hi, 0, sf[0]); stS(srow, n, hi, 16, sf[1]);
    stS(srow, n, hi, 32, sf[2]); stS(srow, n, hi, 48, sf[3]);
    __syncthreads();

    // ---- phase 4: 64 -> 48 ----
    v16h ag0 = ldA(srow, n, hi, 0);
    v16h ag1 = ldA(srow, n, hi, 32);
    v8f sg[3];
#pragma unroll
    for (int nc = 0; nc < 3; ++nc) {
      acc = bias_frag(b_gv[nc]);
      acc = wmma_f16(ag0, ldB(wlds, 21 + nc * 2 + 0, lane), acc);
      acc = wmma_f16(ag1, ldB(wlds, 21 + nc * 2 + 1, lane), acc);
      sg[nc] = lif_frag(m_g[nc], acc);
    }
    __syncthreads();
    stS(srow, n, hi, 0, sg[0]); stS(srow, n, hi, 16, sg[1]);
    stS(srow, n, hi, 32, sg[2]);
    __syncthreads();

    // ---- phase 5: 48 -> 4 (stale k=48..63 killed by zero weights) ----
    v16h ao0 = ldA(srow, n, hi, 0);
    v16h ao1 = ldA(srow, n, hi, 32);
    acc = wmma_f16(ao0, ldB(wlds, 27, lane), bias_frag(b_o));
    acc = wmma_f16(ao1, ldB(wlds, 28, lane), acc);
    v8f so = lif_frag(m_o, acc);

    if (n < 4) {
      size_t ob = ((size_t)t * BATCH + (size_t)m0) * 4 + (size_t)n;
#pragma unroll
      for (int i = 0; i < 8; ++i)
        out[ob + (size_t)(i + 8 * hi) * 4] = so[i];
    }
    __syncthreads();
  }
}

extern "C" void kernel_launch(void* const* d_in, const int* in_sizes, int n_in,
                              void* d_out, int out_size, void* d_ws, size_t ws_size,
                              hipStream_t stream) {
  (void)n_in; (void)out_size; (void)d_ws; (void)ws_size;
  const float* L   = (const float*)d_in[0];
  const float* R   = (const float*)d_in[1];
  const float* Cx  = (const float*)d_in[2];
  const float* Wl1 = (const float*)d_in[3];
  const float* bl1 = (const float*)d_in[4];
  const float* Wl2 = (const float*)d_in[5];
  const float* bl2 = (const float*)d_in[6];
  const float* Wr1 = (const float*)d_in[7];
  const float* br1 = (const float*)d_in[8];
  const float* Wr2 = (const float*)d_in[9];
  const float* br2 = (const float*)d_in[10];
  const float* Wc  = (const float*)d_in[11];
  const float* bc  = (const float*)d_in[12];
  const float* Wf  = (const float*)d_in[13];
  const float* bf  = (const float*)d_in[14];
  const float* Wf2 = (const float*)d_in[15];
  const float* bf2 = (const float*)d_in[16];
  const float* Wo  = (const float*)d_in[17];
  const float* bo  = (const float*)d_in[18];

  const int T = in_sizes[0] / (BATCH * 6);
  dim3 grid(BATCH / (TILE_M * WAVES));   // 512 batch tiles / 4 waves per block
  dim3 block(WAVES * 32);

  hipLaunchKernelGGL(snn_lif_wmma_kernel, grid, block, 0, stream,
                     L, R, Cx, Wl1, bl1, Wl2, bl2, Wr1, br1, Wr2, br2,
                     Wc, bc, Wf, bf, Wf2, bf2, Wo, bo, (float*)d_out, T);
}
